// NTXEntLoss_24275155156992
// MI455X (gfx1250) — matrix operand order repaired and checked by reading, not compiled, not run on hardware
//
#include <hip/hip_runtime.h>
#include <cstdint>
#include <cstddef>

// ---------------- problem constants (reference: N=4096, D=256, T=0.5) ------
#define DCONST   256
#define NCONST   4096
#define TWO_N    (2 * NCONST)
#define NJT      (TWO_N / 16)      // 512 column tiles of 16
#define JSPLIT   8                 // j-range split per i-block (parallelism)
#define JT_PER   (NJT / JSPLIT)    // 64 j-tiles per block
#define WAVES    8                 // waves per block in the GEMM kernel
#define ROWS_PB  (WAVES * 16)      // 128 rows per block
#define IBLOCKS  (NCONST / ROWS_PB)// 32
#define INV_T    2.0f              // 1 / TEMPERATURE

typedef __bf16 bf16_t;
typedef __bf16 v16bf __attribute__((ext_vector_type(16)));
typedef __bf16 v8bf  __attribute__((ext_vector_type(8)));
typedef float  v8f   __attribute__((ext_vector_type(8)));
typedef float  v4f   __attribute__((ext_vector_type(4)));

// ---- async global->LDS copy (CDNA5 ASYNCcnt path), 16 bytes per lane ------
__device__ __forceinline__ void async_cp16(void* lds_dst, const void* gsrc) {
    uint32_t loff = (uint32_t)(uintptr_t)lds_dst;   // low 32 bits = LDS offset
    asm volatile("global_load_async_to_lds_b128 %0, %1, off"
                 :: "v"(loff), "v"(gsrc) : "memory");
}

#define SHUF16(a, b) __builtin_shufflevector((a), (b), 0,1,2,3,4,5,6,7,8,9,10,11,12,13,14,15)

// ============================================================================
// Kernel 1: row L2-normalize; emit bf16 hi/lo split (x = hi + lo ~ fp32)
// One wave per row of the concatenated (2N x D) embedding matrix.
// ============================================================================
__global__ void ntx_normalize(const float* __restrict__ be,
                              const float* __restrict__ ba,
                              bf16_t* __restrict__ ehi,
                              bf16_t* __restrict__ elo) {
    const int lane = threadIdx.x & 31;
    const int wave = threadIdx.x >> 5;
    const int row  = blockIdx.x * WAVES + wave;          // 0 .. 2N-1

    const float* src = (row < NCONST) ? (be + (size_t)row * DCONST)
                                      : (ba + (size_t)(row - NCONST) * DCONST);
    v4f x0 = *(const v4f*)(src + lane * 8);
    v4f x1 = *(const v4f*)(src + lane * 8 + 4);

    float ss = 0.f;
#pragma unroll
    for (int e = 0; e < 4; ++e) ss += x0[e] * x0[e] + x1[e] * x1[e];
    // wave32 reduction
    ss += __shfl_xor(ss, 1);
    ss += __shfl_xor(ss, 2);
    ss += __shfl_xor(ss, 4);
    ss += __shfl_xor(ss, 8);
    ss += __shfl_xor(ss, 16);

    const float scale = 1.0f / fmaxf(sqrtf(ss), 1e-8f);

    v8bf hv, lv;
#pragma unroll
    for (int e = 0; e < 8; ++e) {
        float y = ((e < 4) ? x0[e] : x1[e - 4]) * scale;
        bf16_t h = (bf16_t)y;
        hv[e] = h;
        lv[e] = (bf16_t)(y - (float)h);
    }
    *(v8bf*)(ehi + (size_t)row * DCONST + lane * 8) = hv;
    *(v8bf*)(elo + (size_t)row * DCONST + lane * 8) = lv;
}

// ============================================================================
// Kernel 2: masked log-sum-exp of sim = (emb_n @ emb_n^T) / T via bf16x3 WMMA.
// Block = 8 waves; wave w owns i-tile rows [i0, i0+16); block covers a 64-wide
// slice of the 512 j-tiles. B tiles (16 x 256, hi+lo) double-buffered in LDS
// via async loads; A tiles pinned in VGPRs (wave32 -> big register budget).
// ============================================================================
__global__ void __launch_bounds__(256, 1)
ntx_simloss(const bf16_t* __restrict__ ehi, const bf16_t* __restrict__ elo,
            float* __restrict__ pos, float* __restrict__ partial) {
    // 264 = 256 + 8 bf16 padding -> 16B-aligned rows, no DS bank conflicts
    __shared__ bf16_t btile[2][2][16 * 264];

    const int lane = threadIdx.x & 31;
    const int wave = threadIdx.x >> 5;
    const int ib   = blockIdx.x / JSPLIT;
    const int js   = blockIdx.x % JSPLIT;
    const int jt0  = js * JT_PER;
    const int jt1  = jt0 + JT_PER;
    const int i0   = ib * ROWS_PB + wave * 16;

    // ---- A tile (16 rows x 256 cols, hi/lo) -> registers, WMMA A layout ----
    // lane L<16: row i0+L, K = kc*32 + {0..7, 16..23}
    // lane L>=16: row i0+L-16, K = kc*32 + {8..15, 24..31}
    const int arow = i0 + (lane & 15);
    const int kb   = 8 * (lane >> 4);
    v16bf Ah[8], Al[8];
#pragma unroll
    for (int kc = 0; kc < 8; ++kc) {
        const bf16_t* p = ehi + (size_t)arow * DCONST + kc * 32 + kb;
        v8bf a0 = *(const v8bf*)p;
        v8bf a1 = *(const v8bf*)(p + 16);
        Ah[kc] = SHUF16(a0, a1);
        const bf16_t* q = elo + (size_t)arow * DCONST + kc * 32 + kb;
        v8bf c0 = *(const v8bf*)q;
        v8bf c1 = *(const v8bf*)(q + 16);
        Al[kc] = SHUF16(c0, c1);
    }

    v8f sumacc = {};

    // ---- cooperative B-tile prefetch: 256 threads x 32B x (hi+lo) ----------
    const int trow = threadIdx.x >> 4;   // 0..15 (tile row = column j0+trow)
    const int tseg = threadIdx.x & 15;   // 0..15 (16 bf16 = 32B segment)
    auto prefetch = [&](int jt, int buf) {
        const int j0 = jt * 16;
        const bf16_t* ghi = ehi + (size_t)(j0 + trow) * DCONST + tseg * 16;
        const bf16_t* glo = elo + (size_t)(j0 + trow) * DCONST + tseg * 16;
        bf16_t* dhi = &btile[buf][0][trow * 264 + tseg * 16];
        bf16_t* dlo = &btile[buf][1][trow * 264 + tseg * 16];
        async_cp16(dhi,     ghi);
        async_cp16(dhi + 8, ghi + 8);
        async_cp16(dlo,     glo);
        async_cp16(dlo + 8, glo + 8);
    };

    prefetch(jt0, jt0 & 1);

    for (int jt = jt0; jt < jt1; ++jt) {
        if (jt + 1 < jt1) {
            prefetch(jt + 1, (jt + 1) & 1);
            // 4 newest async ops (next tile) may stay outstanding
            asm volatile("s_wait_asynccnt 4" ::: "memory");
        } else {
            asm volatile("s_wait_asynccnt 0" ::: "memory");
        }
        __syncthreads();

        const int buf  = jt & 1;
        // B layout: lane L<16: col N=L, K=kc*32+{0..7,16..23} ; L>=16: col L-16, K=+{8..15,24..31}
        const int brow = lane & 15;
        const int bkb  = 8 * (lane >> 4);

        v8f acc = {};
#pragma unroll
        for (int kc = 0; kc < 8; ++kc) {
            const bf16_t* ph = &btile[buf][0][brow * 264 + kc * 32 + bkb];
            v8bf h0 = *(const v8bf*)ph;
            v8bf h1 = *(const v8bf*)(ph + 16);
            v16bf Bh = SHUF16(h0, h1);
            const bf16_t* pl = &btile[buf][1][brow * 264 + kc * 32 + bkb];
            v8bf l0 = *(const v8bf*)pl;
            v8bf l1 = *(const v8bf*)(pl + 16);
            v16bf Bl = SHUF16(l0, l1);

            // bf16x3: hi*hi + hi*lo + lo*hi  (drop lo*lo)
            acc = __builtin_amdgcn_wmma_f32_16x16x32_bf16(false, Ah[kc], false, Bh,
                                                          (short)0, acc, false, false);
            acc = __builtin_amdgcn_wmma_f32_16x16x32_bf16(false, Ah[kc], false, Bl,
                                                          (short)0, acc, false, false);
            acc = __builtin_amdgcn_wmma_f32_16x16x32_bf16(false, Al[kc], false, Bh,
                                                          (short)0, acc, false, false);
        }

        // C layout: element r of lane L -> row M = r + 8*(L>=16), col N = L%16
        const int gj     = jt * 16 + (lane & 15);
        const int gibase = i0 + 8 * (lane >> 4);
#pragma unroll
        for (int r = 0; r < 8; ++r) {
            float sim = acc[r] * INV_T;
            int   gi  = gibase + r;                       // gi < N
            if ((gj & (NCONST - 1)) != gi) sumacc[r] += __expf(sim);
            if (gj == gi + NCONST) pos[gi] = sim;         // unique writer
        }
        __syncthreads();   // compute done before next overwrite of this buffer
    }

    // reduce the 16 columns inside each lane-half (masks stay within halves)
#pragma unroll
    for (int r = 0; r < 8; ++r) {
        float v = sumacc[r];
        v += __shfl_xor(v, 1);
        v += __shfl_xor(v, 2);
        v += __shfl_xor(v, 4);
        v += __shfl_xor(v, 8);
        sumacc[r] = v;
    }
    if ((lane & 15) == 0) {
        const int gibase = i0 + 8 * (lane >> 4);
#pragma unroll
        for (int r = 0; r < 8; ++r)
            partial[(size_t)js * NCONST + gibase + r] = sumacc[r];
    }
}

// ============================================================================
// Kernel 3: loss = mean( log(sum_exp_i) - pos_i )
// ============================================================================
__global__ void ntx_finalize(const float* __restrict__ pos,
                             const float* __restrict__ partial,
                             float* __restrict__ out) {
    __shared__ float red[256];
    float acc = 0.f;
    for (int i = threadIdx.x; i < NCONST; i += 256) {
        float s = 0.f;
#pragma unroll
        for (int js = 0; js < JSPLIT; ++js) s += partial[(size_t)js * NCONST + i];
        acc += logf(s) - pos[i];
    }
    red[threadIdx.x] = acc;
    __syncthreads();
    for (int st = 128; st > 0; st >>= 1) {
        if (threadIdx.x < st) red[threadIdx.x] += red[threadIdx.x + st];
        __syncthreads();
    }
    if (threadIdx.x == 0) out[0] = red[0] / (float)NCONST;
}

// ============================================================================
extern "C" void kernel_launch(void* const* d_in, const int* in_sizes, int n_in,
                              void* d_out, int out_size, void* d_ws, size_t ws_size,
                              hipStream_t stream) {
    (void)in_sizes; (void)n_in; (void)out_size; (void)ws_size;
    const float* be = (const float*)d_in[0];
    const float* ba = (const float*)d_in[1];

    char* ws = (char*)d_ws;
    bf16_t* ehi     = (bf16_t*)ws;                                       // 4 MB
    bf16_t* elo     = (bf16_t*)(ws + (size_t)TWO_N * DCONST * 2);        // 4 MB
    float*  pos     = (float*) (ws + (size_t)TWO_N * DCONST * 4);        // 16 KB
    float*  partial = (float*) (ws + (size_t)TWO_N * DCONST * 4
                                   + (size_t)NCONST * sizeof(float));    // 128 KB
    float* out = (float*)d_out;

    ntx_normalize<<<TWO_N / WAVES, 256, 0, stream>>>(be, ba, ehi, elo);
    ntx_simloss<<<IBLOCKS * JSPLIT, 256, 0, stream>>>(ehi, elo, pos, partial);
    ntx_finalize<<<1, 256, 0, stream>>>(pos, partial, out);
}
